// GRU_39994735460383
// MI455X (gfx1250) — matrix operand order repaired and checked by reading, not compiled, not run on hardware
//
#include <hip/hip_runtime.h>

// ---------------------------------------------------------------------------
// GRU cell, B=131072, I=H=128, fp32 in/out, bf16 WMMA compute (CDNA5 gfx1250).
//
// Layouts (per CDNA5 ISA 7.12.2, wave32):
//   A 16x32 bf16 frag, element e of v16bf:  k = (e/8)*16 + sel*8 + (e%8),
//       row m = lane&15, sel = lane>>4.
//   B 32x16 bf16 frag, element e of v16bf:  k = (e/8)*16 + sel*8 + (e%8), col n = lane&15.
//   C/D 16x16 f32 frag, element i of v8f:   row m = sel*8 + i, col n = lane&15.
// ---------------------------------------------------------------------------

typedef __attribute__((ext_vector_type(16))) __bf16 v16bf;
typedef __attribute__((ext_vector_type(8)))  __bf16 v8bf;
typedef __attribute__((ext_vector_type(8)))  float  v8f;
typedef __attribute__((ext_vector_type(4)))  float  v4f;

#define KS    136                  // padded K stride (bf16 elems) -> 272B rows, conflict-free
#define WMAT  (128 * KS)           // one 128x128 weight matrix in LDS (bf16 elems)
#define SA_WAVE (16 * KS)          // per-wave r*h staging tile (bf16 elems)
#define LDS_ELEMS (6 * WMAT + 4 * SA_WAVE)

__device__ __forceinline__ v8f wmma_bf16(v16bf a, v16bf b, v8f c) {
    return __builtin_amdgcn_wmma_f32_16x16x32_bf16(
        /*neg_a=*/false, a, /*neg_b=*/false, b,
        /*c_mod=*/(short)0, c, /*reuse_a=*/false, /*reuse_b=*/false);
}

// A-fragment from a global fp32 row. p -> &x[row*128 + kt*32 + sel*8].
__device__ __forceinline__ v16bf load_a_global(const float* __restrict__ p) {
    v4f a0 = *(const v4f*)(p + 0);
    v4f a1 = *(const v4f*)(p + 4);
    v4f b0 = *(const v4f*)(p + 16);
    v4f b1 = *(const v4f*)(p + 20);
    v16bf r;
#pragma unroll
    for (int j = 0; j < 4; ++j) {
        r[j + 0]  = (__bf16)a0[j];
        r[j + 4]  = (__bf16)a1[j];
        r[j + 8]  = (__bf16)b0[j];
        r[j + 12] = (__bf16)b1[j];
    }
    return r;
}

// B-fragment from LDS [N][K] weight tile. p -> &mat[n*KS + sel*8 + kt*32].
__device__ __forceinline__ v16bf load_b_lds(const __bf16* p) {
    v8bf lo = *(const v8bf*)(p);
    v8bf hi = *(const v8bf*)(p + 16);
    return __builtin_shufflevector(lo, hi, 0,1,2,3,4,5,6,7,8,9,10,11,12,13,14,15);
}

// A-fragment from LDS [M][K] staging tile. p -> &sA[m*KS + kt*32 + sel*8].
__device__ __forceinline__ v16bf load_a_lds(const __bf16* p) {
    v8bf lo = *(const v8bf*)(p);
    v8bf hi = *(const v8bf*)(p + 16);
    return __builtin_shufflevector(lo, hi, 0,1,2,3,4,5,6,7,8,9,10,11,12,13,14,15);
}

__device__ __forceinline__ float sigmoid_f(float x) {
    return 1.0f / (1.0f + __expf(-x));
}
__device__ __forceinline__ float tanh_f(float x) {
    return 1.0f - 2.0f / (__expf(2.0f * x) + 1.0f);
}

__global__ __launch_bounds__(128)
void gru_fused_wmma(const float* __restrict__ x,
                    const float* __restrict__ h,
                    const float* __restrict__ Wz, const float* __restrict__ Uz,
                    const float* __restrict__ bz,
                    const float* __restrict__ Wr, const float* __restrict__ Ur,
                    const float* __restrict__ br,
                    const float* __restrict__ Wh, const float* __restrict__ Uh,
                    const float* __restrict__ bh,
                    float* __restrict__ out)
{
    extern __shared__ __bf16 smem[];

    const int tid = threadIdx.x;

    // ---- Stage all six weight matrices into LDS, bf16, transposed [N][K]. ----
    // Thread t owns global row k=t of each matrix: reads are contiguous per
    // thread; LDS writes land in bank (t>>1)&63 -> only pairwise conflicts.
    {
        const float* mats[6] = { Wz, Uz, Wr, Ur, Wh, Uh };
#pragma unroll
        for (int mi = 0; mi < 6; ++mi) {
            const float* __restrict__ G = mats[mi] + tid * 128;
            __bf16* dst = smem + mi * WMAT;
#pragma unroll
            for (int n4 = 0; n4 < 32; ++n4) {
                v4f v = *(const v4f*)(G + n4 * 4);
#pragma unroll
                for (int j = 0; j < 4; ++j)
                    dst[(n4 * 4 + j) * KS + tid] = (__bf16)v[j];
            }
        }
    }
    __syncthreads();

    const int lane = tid & 31;
    const int wave = tid >> 5;
    const int mlo  = lane & 15;
    const int sel  = lane >> 4;
    const long r0  = (long)blockIdx.x * 64 + (long)wave * 16;

    // ---- A-fragments for x and h_prev rows (K = 128 -> 4 k-tiles). ----
    const float* __restrict__ xrow = x + (r0 + mlo) * 128;
    const float* __restrict__ hrow = h + (r0 + mlo) * 128;
    v16bf Ax[4], Ah[4];
#pragma unroll
    for (int kt = 0; kt < 4; ++kt) {
        Ax[kt] = load_a_global(xrow + kt * 32 + sel * 8);
        Ah[kt] = load_a_global(hrow + kt * 32 + sel * 8);
    }

    v8f accz[8], accr[8], acch[8];
#pragma unroll
    for (int nt = 0; nt < 8; ++nt)
#pragma unroll
        for (int i = 0; i < 8; ++i) { accz[nt][i] = 0.f; accr[nt][i] = 0.f; acch[nt][i] = 0.f; }

    // ---- Phase 1 GEMMs: accz = x@Wz + h@Uz, accr = x@Wr + h@Ur, acch = x@Wh.
#pragma unroll
    for (int nt = 0; nt < 8; ++nt) {
        const int nbase = (nt * 16 + mlo) * KS + sel * 8;
#pragma unroll
        for (int kt = 0; kt < 4; ++kt) {
            const int kb = kt * 32;
            accz[nt] = wmma_bf16(Ax[kt], load_b_lds(smem + 0 * WMAT + nbase + kb), accz[nt]);
            accz[nt] = wmma_bf16(Ah[kt], load_b_lds(smem + 1 * WMAT + nbase + kb), accz[nt]);
            accr[nt] = wmma_bf16(Ax[kt], load_b_lds(smem + 2 * WMAT + nbase + kb), accr[nt]);
            accr[nt] = wmma_bf16(Ah[kt], load_b_lds(smem + 3 * WMAT + nbase + kb), accr[nt]);
            acch[nt] = wmma_bf16(Ax[kt], load_b_lds(smem + 4 * WMAT + nbase + kb), acch[nt]);
        }
    }

    // ---- Gates: z = sig(accz+bz) (kept in accz), a = sig(accr+br)*h -> LDS.
    __bf16* sA = smem + 6 * WMAT + wave * SA_WAVE;   // private per-wave tile
    v8f hC[8];
    const long rowb = r0 + sel * 8;
#pragma unroll
    for (int nt = 0; nt < 8; ++nt) {
        const int col = nt * 16 + mlo;
        const float bzv = bz[col];
        const float brv = br[col];
#pragma unroll
        for (int i = 0; i < 8; ++i) {
            const float zv = sigmoid_f(accz[nt][i] + bzv);
            const float rv = sigmoid_f(accr[nt][i] + brv);
            const float hv = h[(rowb + i) * 128 + col];
            hC[nt][i]   = hv;
            accz[nt][i] = zv;                               // keep z for final blend
            sA[(sel * 8 + i) * KS + col] = (__bf16)(rv * hv);
        }
    }
    // Same-wave LDS RAW: drain DS counter before reading the staging tile back.
    asm volatile("s_wait_dscnt 0x0" ::: "memory");

    // ---- Phase 2 GEMM: acch += (r*h) @ Uh. ----
    v16bf Aa[4];
#pragma unroll
    for (int kt = 0; kt < 4; ++kt)
        Aa[kt] = load_a_lds(sA + mlo * KS + kt * 32 + sel * 8);

#pragma unroll
    for (int nt = 0; nt < 8; ++nt) {
        const int nbase = (nt * 16 + mlo) * KS + sel * 8;
#pragma unroll
        for (int kt = 0; kt < 4; ++kt)
            acch[nt] = wmma_bf16(Aa[kt], load_b_lds(smem + 5 * WMAT + nbase + kt * 32), acch[nt]);
    }

    // ---- h_t = (1-z)*h + z*tanh(acch + bh) ----
#pragma unroll
    for (int nt = 0; nt < 8; ++nt) {
        const int col = nt * 16 + mlo;
        const float bhv = bh[col];
#pragma unroll
        for (int i = 0; i < 8; ++i) {
            const float hc = tanh_f(acch[nt][i] + bhv);
            const float zv = accz[nt][i];
            out[(rowb + i) * 128 + col] = (1.0f - zv) * hC[nt][i] + zv * hc;
        }
    }
}

extern "C" void kernel_launch(void* const* d_in, const int* in_sizes, int n_in,
                              void* d_out, int out_size, void* d_ws, size_t ws_size,
                              hipStream_t stream) {
    const float* x  = (const float*)d_in[0];
    const float* h  = (const float*)d_in[1];
    const float* Wz = (const float*)d_in[2];
    const float* Uz = (const float*)d_in[3];
    const float* bz = (const float*)d_in[4];
    const float* Wr = (const float*)d_in[5];
    const float* Ur = (const float*)d_in[6];
    const float* br = (const float*)d_in[7];
    const float* Wh = (const float*)d_in[8];
    const float* Uh = (const float*)d_in[9];
    const float* bh = (const float*)d_in[10];
    float* out = (float*)d_out;

    const int B = in_sizes[0] / 128;          // 131072
    const int blocks = B / 64;                // 64 rows per 4-wave workgroup
    const size_t lds_bytes = (size_t)LDS_ELEMS * 2;   // bf16 elems * 2B ≈ 221 KB

    gru_fused_wmma<<<dim3(blocks), dim3(128), lds_bytes, stream>>>(
        x, h, Wz, Uz, bz, Wr, Ur, br, Wh, Uh, bh, out);
}
